// OneStepGenerator_88012469829725
// MI455X (gfx1250) — matrix operand order repaired and checked by reading, not compiled, not run on hardware
//
#include <hip/hip_runtime.h>
#include <hip/hip_bf16.h>
#include <math.h>

// ---------------- Problem constants (from reference) ----------------
#define ENC_DIM 1024
#define TOK_DIM 256
#define HIDDEN  1024
#define BATCH   512
#define CANDS   2048
#define TOPK    64

// GEMM staging parameters
#define KSTEP 64              // K elements staged per async stage
#define AROW  72              // padded LDS row length in bf16 (144 B -> conflict-free frag reads)

// ---------------- WMMA vector types (CDNA5 / gfx1250, wave32) --------
typedef __attribute__((ext_vector_type(8)))  __bf16 v8bf;
typedef __attribute__((ext_vector_type(16))) __bf16 v16bf;
typedef __attribute__((ext_vector_type(8)))  float  v8f;

// ---------------- f32 -> bf16 conversion kernels ---------------------
__global__ void osg_cvt_bf16(const float* __restrict__ in, __bf16* __restrict__ out, int n) {
    int i = blockIdx.x * blockDim.x + threadIdx.x;
    if (i < n) out[i] = (__bf16)in[i];
}

// in is [K][N] row-major f32; out is [N][K] row-major bf16 (transposed)
__global__ void osg_cvt_bf16_t(const float* __restrict__ in, __bf16* __restrict__ out,
                               int K, int N) {
    int i = blockIdx.x * blockDim.x + threadIdx.x;
    if (i < K * N) {
        int k = i / N;
        int n = i - k * N;
        out[(size_t)n * K + k] = (__bf16)in[i];
    }
}

// ---------------- WMMA GEMM: D = act(A @ B + bias) -------------------
// A: [M x Kd] bf16 row-major.  Bt: [N x Kd] bf16 (B transposed, so each
// output column's K-run is contiguous).  Block = 4 waves; all waves share
// one 16-row A panel, double-buffered in LDS via GLOBAL_LOAD_ASYNC_TO_LDS
// (ASYNCcnt pipeline); each wave computes its own 16x16 N-subtile.
// Fragment layouts follow ISA 7.12.2 (16-bit A 16x32, B 32x16, f32 C/D).
__global__ void osg_gemm_wmma_bf16(const __bf16* __restrict__ A,
                                   const __bf16* __restrict__ Bt,
                                   const float*  __restrict__ bias,
                                   float*        __restrict__ outF,   // f32 output (or null)
                                   __bf16*       __restrict__ outB,   // bf16 output (or null)
                                   int M, int N, int Kd, int doGelu) {
    __shared__ __bf16 aLds[2][16 * AROW];       // 2 x 2304 B double buffer

    const int tid  = threadIdx.x;
    const int lane = tid & 31;
    const int wave = tid >> 5;
    const int rowBase = blockIdx.y * 16;
    const int colBase = (blockIdx.x * 4 + wave) * 16;

    const int l16   = lane & 15;
    const int lhalf = lane >> 4;          // 0: lanes 0-15, 1: lanes 16-31
    const int bn    = colBase + l16;      // B column owned by this lane
    const int akoff = lhalf * 8;          // A frag: K chunks at akoff and 16+akoff
    const int bkoff = lhalf * 8;          // B frag: K chunks at bkoff and 16+bkoff

    // Async-copy assignment: 128 threads x 16 B = 2 KB = 16 rows x 64 bf16.
    const int cpR = tid >> 3;                         // row 0..15
    const int cpC = (tid & 7) * 8;                    // bf16 column chunk
    const __bf16* gsrc = A + (size_t)(rowBase + cpR) * Kd + cpC;
    // LDS byte offsets (low 32 bits of a generic LDS pointer == allocation offset)
    const unsigned ldsDst0 = (unsigned)(size_t)(&aLds[0][cpR * AROW + cpC]);
    const unsigned ldsDst1 = (unsigned)(size_t)(&aLds[1][cpR * AROW + cpC]);

    const __bf16* bRow = Bt + (size_t)bn * Kd;
    const int nst = Kd / KSTEP;
    v8f acc = {};

    // Prologue: issue stage 0 into buffer 0.
    asm volatile("global_load_async_to_lds_b128 %0, %1, off"
                 :: "v"(ldsDst0), "v"(gsrc) : "memory");

    for (int s = 0; s < nst; ++s) {
        if (s + 1 < nst) {
            // Issue stage s+1 into the other buffer, then retire stage s
            // (async loads complete in order; <=1 leaves only stage s+1 live).
            const __bf16* gnext = gsrc + (s + 1) * KSTEP;
            unsigned dnext = ((s + 1) & 1) ? ldsDst1 : ldsDst0;
            asm volatile("global_load_async_to_lds_b128 %0, %1, off"
                         :: "v"(dnext), "v"(gnext) : "memory");
            asm volatile("s_wait_asynccnt 0x1" ::: "memory");
        } else {
            asm volatile("s_wait_asynccnt 0x0" ::: "memory");
        }
        __syncthreads();                   // stage s visible to all 4 waves

        const __bf16* aBuf = &aLds[s & 1][0];
        const int kg = s * KSTEP;          // global K base of this stage
#pragma unroll
        for (int kk = 0; kk < KSTEP; kk += 32) {
            const __bf16* ap = aBuf + l16 * AROW + kk + akoff;
            v8bf a0 = *(const v8bf*)(ap);          // K = kk+akoff .. +7
            v8bf a1 = *(const v8bf*)(ap + 16);     // K = kk+16+akoff .. +7
            v16bf afrag = __builtin_shufflevector(a0, a1,
                            0,1,2,3,4,5,6,7,8,9,10,11,12,13,14,15);

            const __bf16* bp = bRow + kg + kk + bkoff;
            v8bf b0 = *(const v8bf*)(bp);          // K = kk+bkoff .. +7
            v8bf b1 = *(const v8bf*)(bp + 16);     // K = kk+16+bkoff .. +7
            v16bf bfrag = __builtin_shufflevector(b0, b1,
                            0,1,2,3,4,5,6,7,8,9,10,11,12,13,14,15);

            // (neg_a, A, neg_b, B, c_mod, C, reuse_a, reuse_b)
            acc = __builtin_amdgcn_wmma_f32_16x16x32_bf16(
                      false, afrag, false, bfrag, (short)0, acc, false, false);
        }
        __syncthreads();   // all waves done reading buf (s&1) before stage s+2 overwrites it
    }

    // C/D layout: VGPR r, lanes 0-15 -> M=r, N=lane; lanes 16-31 -> M=r+8.
    const int n = colBase + l16;
    float bv = bias ? bias[n] : 0.0f;
#pragma unroll
    for (int r = 0; r < 8; ++r) {
        int m = rowBase + r + lhalf * 8;
        float v = acc[r] + bv;
        if (doGelu) v = 0.5f * v * (1.0f + erff(v * 0.70710678118654752f));
        if (outB) outB[(size_t)m * N + n] = (__bf16)v;
        else      outF[(size_t)m * N + n] = v;
    }
}

// ---------------- logits[b,c] = dot(q[b], tok_emb[cand[b,c]]) --------
// One block per batch row. q[b] (1 KB) staged in LDS; each thread walks
// whole gathered rows with float4 loads (tok_emb is L2-resident: 102 MB
// vs 192 MB L2 -> this kernel is L2-bandwidth bound, WMMA not applicable
// since every (b,c) pair gathers a distinct row).
__global__ void osg_logits(const float* __restrict__ q,
                           const float* __restrict__ emb,
                           const int*   __restrict__ cand,
                           float*       __restrict__ logits) {
    const int b = blockIdx.x;
    __shared__ float qs[TOK_DIM];
    for (int i = threadIdx.x; i < TOK_DIM; i += blockDim.x)
        qs[i] = q[(size_t)b * TOK_DIM + i];
    __syncthreads();

    for (int c = threadIdx.x; c < CANDS; c += blockDim.x) {
        int tok = cand[(size_t)b * CANDS + c];
        const float4* row = (const float4*)(emb + (size_t)tok * TOK_DIM);
        float acc = 0.0f;
#pragma unroll 8
        for (int k = 0; k < TOK_DIM / 4; ++k) {
            float4 v = row[k];
            acc += v.x * qs[4 * k + 0] + v.y * qs[4 * k + 1] +
                   v.z * qs[4 * k + 2] + v.w * qs[4 * k + 3];
        }
        logits[(size_t)b * CANDS + c] = (tok < 2) ? -INFINITY : acc;
    }
}

// ---------------- top-64 per row (stable: smallest index on ties) -----
__global__ void osg_topk(const float* __restrict__ logits,
                         const int*   __restrict__ cand,
                         int*         __restrict__ outTok) {
    const int b   = blockIdx.x;
    const int tid = threadIdx.x;
    __shared__ float vals[CANDS];      // 8 KB
    __shared__ float sVal[256];
    __shared__ int   sIdx[256];

    for (int i = tid; i < CANDS; i += blockDim.x)
        vals[i] = logits[(size_t)b * CANDS + i];
    __syncthreads();

    for (int t = 0; t < TOPK; ++t) {
        float best = -INFINITY;
        int   bi   = 0x7fffffff;
        for (int i = tid; i < CANDS; i += blockDim.x) {
            float v = vals[i];
            if (v > best || (v == best && i < bi)) { best = v; bi = i; }
        }
        sVal[tid] = best; sIdx[tid] = bi;
        __syncthreads();
        for (int s = 128; s > 0; s >>= 1) {
            if (tid < s) {
                float ov = sVal[tid + s]; int oi = sIdx[tid + s];
                if (ov > sVal[tid] || (ov == sVal[tid] && oi < sIdx[tid])) {
                    sVal[tid] = ov; sIdx[tid] = oi;
                }
            }
            __syncthreads();
        }
        if (tid == 0) {
            int w = sIdx[0];
            outTok[(size_t)b * TOPK + t] = cand[(size_t)b * CANDS + w];
            vals[w] = -INFINITY;
        }
        __syncthreads();
    }
}

// ---------------- launcher ------------------------------------------
extern "C" void kernel_launch(void* const* d_in, const int* in_sizes, int n_in,
                              void* d_out, int out_size, void* d_ws, size_t ws_size,
                              hipStream_t stream) {
    const float* enc  = (const float*)d_in[0];   // [512,1024]
    const int*   cand = (const int*)  d_in[1];   // [512,2048]
    // d_in[2] = topk scalar (== 64, fixed by the problem)
    const float* emb  = (const float*)d_in[3];   // [100000,256]
    const float* W1   = (const float*)d_in[4];   // [1024,1024]
    const float* b1   = (const float*)d_in[5];   // [1024]
    const float* W2   = (const float*)d_in[6];   // [1024,256]

    char* ws = (char*)d_ws;
    __bf16* encB = (__bf16*)(ws + 0);                  // 512*1024*2  = 1 MB
    __bf16* w1t  = (__bf16*)(ws + 1048576);            // 1024*1024*2 = 2 MB
    __bf16* w2t  = (__bf16*)(ws + 3145728);            // 256*1024*2  = 0.5 MB
    __bf16* hB   = (__bf16*)(ws + 3670016);            // 512*1024*2  = 1 MB
    float*  qbuf = (float*)(ws + 4718592);             // 512*256*4   = 0.5 MB

    float* logits = (float*)d_out;                           // [512,2048] f32
    int*   outTok = (int*)((float*)d_out + BATCH * CANDS);   // [512,64] int32 bits

    // bf16 staging (W1/W2 transposed so WMMA B fragments are contiguous)
    osg_cvt_bf16  <<<(BATCH * ENC_DIM) / 256, 256, 0, stream>>>(enc, encB, BATCH * ENC_DIM);
    osg_cvt_bf16_t<<<(ENC_DIM * HIDDEN) / 256, 256, 0, stream>>>(W1, w1t, ENC_DIM, HIDDEN);
    osg_cvt_bf16_t<<<(HIDDEN * TOK_DIM) / 256, 256, 0, stream>>>(W2, w2t, HIDDEN, TOK_DIM);

    // h = GELU(enc @ W1 + b1)   [512,1024], bf16 out
    osg_gemm_wmma_bf16<<<dim3(HIDDEN / 16 / 4, BATCH / 16), 128, 0, stream>>>(
        encB, w1t, b1, nullptr, hB, BATCH, HIDDEN, ENC_DIM, 1);

    // q = h @ W2                [512,256], f32 out
    osg_gemm_wmma_bf16<<<dim3(TOK_DIM / 16 / 4, BATCH / 16), 128, 0, stream>>>(
        hB, w2t, nullptr, qbuf, nullptr, BATCH, TOK_DIM, HIDDEN, 0);

    // logits + mask
    osg_logits<<<BATCH, 256, 0, stream>>>(qbuf, emb, cand, logits);

    // top-64 tokens
    osg_topk<<<BATCH, 256, 0, stream>>>(logits, cand, outTok);
}
